// classifier_2972117369050
// MI455X (gfx1250) — matrix operand order repaired and checked by reading, not compiled, not run on hardware
//
#include <hip/hip_runtime.h>
#include <hip/hip_bf16.h>

// Sizes from the reference
#define B_   128
#define S1_  16
#define DQ_  300
#define DQP_ 320     // DQ padded to multiple of 32 for WMMA K-loop
#define C_   256
#define S2_  8
#define D_   256
#define MQ_  2048    // B*S1
#define MS_  2048    // C*S2

typedef __attribute__((ext_vector_type(16))) __bf16 v16bf;
typedef __attribute__((ext_vector_type(8)))  float  v8f;

// ---------------- WMMA fragment loaders (wave32, 16x16x32 bf16) ----------------
// A (16x32, MxK): lane m=lane&15, half h=lane>>4; elems 0..15 <- A[m][kb+16h .. +15]
//                 (one 32B contiguous load)
__device__ __forceinline__ v16bf frag_a(const __bf16* A, int lda, int m0, int kb, int lane) {
  const __bf16* p = A + (size_t)(m0 + (lane & 15)) * lda + kb + ((lane >> 4) * 16);
  v16bf f;
#pragma unroll
  for (int i = 0; i < 8; ++i) { f[i] = p[i]; f[i + 8] = p[8 + i]; }
  return f;
}
// B (32x16, KxN) for NT-GEMM: B[k][n] = Bt[n][k]; lane n=lane&15, half h=lane>>4;
// elems 0..15 <- Bt[n][kb+16h .. +15]  (one 32B contiguous load)
__device__ __forceinline__ v16bf frag_bt(const __bf16* Bt, int ldb, int n0, int kb, int lane) {
  const __bf16* p = Bt + (size_t)(n0 + (lane & 15)) * ldb + kb + ((lane >> 4) * 16);
  v16bf f;
#pragma unroll
  for (int i = 0; i < 16; ++i) f[i] = p[i];
  return f;
}

// ordered-uint transform so float max can use atomicMax(unsigned) deterministically
__device__ __forceinline__ unsigned f2ord(float f) {
  unsigned u = __float_as_uint(f);
  return (u & 0x80000000u) ? ~u : (u | 0x80000000u);
}
__device__ __forceinline__ float ord2f(unsigned o) {
  return (o & 0x80000000u) ? __uint_as_float(o & 0x7fffffffu) : __uint_as_float(~o);
}

// ---------------- conversion kernels ----------------
__global__ void k_cvt_pad(const float* __restrict__ in, __bf16* __restrict__ out,
                          int kin, int kout) {
  int r = blockIdx.x;
  for (int k = threadIdx.x; k < kout; k += blockDim.x)
    out[(size_t)r * kout + k] = (__bf16)(k < kin ? in[(size_t)r * kin + k] : 0.0f);
}
__global__ void k_cvt_T_pad(const float* __restrict__ in /*[kin][n]*/, __bf16* __restrict__ out /*[n][kout]*/,
                            int n, int kin, int kout) {
  int c = blockIdx.x;
  for (int k = threadIdx.x; k < kout; k += blockDim.x)
    out[(size_t)c * kout + k] = (__bf16)(k < kin ? in[(size_t)k * n + c] : 0.0f);
}

// ---- generic NT GEMM: C[M,N] = A[M,K] * Bt[N,K]^T ----
// 2x2 register blocking: one wave computes a 32x32 tile (4 accumulators), so each
// A/B fragment load feeds two WMMAs (2 b128 loads per v_wmma instead of 4).
template <int OUT_BF16>
__global__ void gemm_nt(const __bf16* __restrict__ A, const __bf16* __restrict__ Bt,
                        void* __restrict__ Cout, int ntn2, int K, int lda, int ldb, int ldc) {
  const int lane = threadIdx.x;
  const int m0 = (blockIdx.x / ntn2) * 32;
  const int n0 = (blockIdx.x % ntn2) * 32;
  v8f acc00 = {}, acc01 = {}, acc10 = {}, acc11 = {};
  for (int kb = 0; kb < K; kb += 32) {
    v16bf a0 = frag_a(A, lda, m0,      kb, lane);
    v16bf a1 = frag_a(A, lda, m0 + 16, kb, lane);
    v16bf b0 = frag_bt(Bt, ldb, n0,      kb, lane);
    v16bf b1 = frag_bt(Bt, ldb, n0 + 16, kb, lane);
    acc00 = __builtin_amdgcn_wmma_f32_16x16x32_bf16(false, a0, false, b0, (short)0, acc00, false, false);
    acc01 = __builtin_amdgcn_wmma_f32_16x16x32_bf16(false, a0, false, b1, (short)0, acc01, false, false);
    acc10 = __builtin_amdgcn_wmma_f32_16x16x32_bf16(false, a1, false, b0, (short)0, acc10, false, false);
    acc11 = __builtin_amdgcn_wmma_f32_16x16x32_bf16(false, a1, false, b1, (short)0, acc11, false, false);
  }
  const int nlo = lane & 15;
  const int rb  = (lane >> 4) * 8;
  if (OUT_BF16) {
    __bf16* Cb = (__bf16*)Cout;
#pragma unroll
    for (int r = 0; r < 8; ++r) {
      Cb[(size_t)(m0 + rb + r) * ldc + n0 + nlo]           = (__bf16)acc00[r];
      Cb[(size_t)(m0 + rb + r) * ldc + n0 + 16 + nlo]      = (__bf16)acc01[r];
      Cb[(size_t)(m0 + 16 + rb + r) * ldc + n0 + nlo]      = (__bf16)acc10[r];
      Cb[(size_t)(m0 + 16 + rb + r) * ldc + n0 + 16 + nlo] = (__bf16)acc11[r];
    }
  } else {
    float* Cf = (float*)Cout;
#pragma unroll
    for (int r = 0; r < 8; ++r) {
      Cf[(size_t)(m0 + rb + r) * ldc + n0 + nlo]           = acc00[r];
      Cf[(size_t)(m0 + rb + r) * ldc + n0 + 16 + nlo]      = acc01[r];
      Cf[(size_t)(m0 + 16 + rb + r) * ldc + n0 + nlo]      = acc10[r];
      Cf[(size_t)(m0 + 16 + rb + r) * ldc + n0 + 16 + nlo] = acc11[r];
    }
  }
}

// ---------------- fused scores -> softmax(t) -> mean(s) -> abar ----------------
// Tile rows M = (class pair: c0 -> M 0..7, c1 -> M 8..15) x t; cols N = s (16).
// Each lane holds all 8 t-values of one (class, s): softmax is lane-local over acc[0..7];
// mean over s is a 16-lane shfl reduction.  This tile shape IS the fusion, keep 16x16.
__global__ void k_scores_abar(const __bf16* __restrict__ Kb, const __bf16* __restrict__ Qb,
                              float* __restrict__ abar) {
  const int lane = threadIdx.x;
  const int b  = blockIdx.x >> 7;     // 128 class-pairs per b
  const int cp = blockIdx.x & 127;
  v8f acc = {};
  for (int kb = 0; kb < D_; kb += 32) {
    v16bf a = frag_a(Kb, D_, cp * 16, kb, lane);   // rows c*8+t, contiguous = cp*16..+15
    v16bf q = frag_bt(Qb, D_, b * 16, kb, lane);   // cols = s rows of Q
    acc = __builtin_amdgcn_wmma_f32_16x16x32_bf16(false, a, false, q, (short)0, acc, false, false);
  }
  const float scale = 0.05773502691896258f;        // 1/sqrt(d_k = 300)
  float v[8]; float mx = -3.402823466e38f;
#pragma unroll
  for (int t = 0; t < 8; ++t) { v[t] = acc[t] * scale; mx = fmaxf(mx, v[t]); }
  float sum = 0.f;
#pragma unroll
  for (int t = 0; t < 8; ++t) { v[t] = __expf(v[t] - mx); sum += v[t]; }
  const float inv = 1.0f / sum;
  const int h = lane >> 4;                         // 0 -> class c0, 1 -> class c1
  const int c = cp * 2 + h;
#pragma unroll
  for (int t = 0; t < 8; ++t) {
    float w = v[t] * inv;
    w += __shfl_xor(w, 1); w += __shfl_xor(w, 2);
    w += __shfl_xor(w, 4); w += __shfl_xor(w, 8);  // sum over the 16 s-columns
    if ((lane & 15) == 0)
      abar[((size_t)b * C_ + c) * 8 + t] = w * (1.0f / 16.0f);
  }
}

// ---------------- proto[b,c,:] = abar[b,c,:] @ V[c] ----------------
__global__ void k_proto(const float* __restrict__ abar, const float* __restrict__ V,
                        float* __restrict__ proto) {
  int bc = blockIdx.x, d = threadIdx.x;
  int c = bc & (C_ - 1);
  __shared__ float aw[8];
  if (d < 8) aw[d] = abar[(size_t)bc * 8 + d];
  __syncthreads();
  float s = 0.f;
#pragma unroll
  for (int t = 0; t < 8; ++t) s += aw[t] * V[((size_t)c * 8 + t) * D_ + d];
  proto[(size_t)bc * D_ + d] = s;
}

__global__ void k_center(const float* __restrict__ proto, float* __restrict__ center) {
  int b = blockIdx.x, d = threadIdx.x;
  float s = 0.f;
  for (int c = 0; c < C_; ++c) s += proto[((size_t)b * C_ + c) * D_ + d];
  center[b * D_ + d] = s * (1.0f / C_);
}

__global__ void k_logits(const float* __restrict__ left, const float* __restrict__ proto,
                         float* __restrict__ logits) {
  int b = blockIdx.x, c = threadIdx.x;
  __shared__ float lo[D_];
  lo[c] = left[b * D_ + c];
  __syncthreads();
  const float* pr = proto + ((size_t)b * C_ + c) * D_;
  float s = 0.f;
  for (int d = 0; d < D_; ++d) s += lo[d] * pr[d];
  logits[b * C_ + c] = s * (1.0f / 0.9f);          // /TEMPERATURE
}

__global__ void k_ce(const float* __restrict__ logits, const int* __restrict__ ids,
                     float* __restrict__ ceb) {
  int b = blockIdx.x, lane = threadIdx.x;
  float mx = -3.402823466e38f;
  for (int c = lane; c < C_; c += 32) mx = fmaxf(mx, logits[b * C_ + c]);
  for (int m = 1; m < 32; m <<= 1) mx = fmaxf(mx, __shfl_xor(mx, m));
  float s = 0.f;
  for (int c = lane; c < C_; c += 32) s += __expf(logits[b * C_ + c] - mx);
  for (int m = 1; m < 32; m <<= 1) s += __shfl_xor(s, m);
  if (lane == 0) ceb[b] = (mx + __logf(s)) - logits[b * C_ + ids[b]];
}

// ---------------- normalize(proto - center) -> bf16 npe ----------------
__global__ void k_npe(const float* __restrict__ proto, const float* __restrict__ center,
                      __bf16* __restrict__ npe) {
  int bc = blockIdx.x, lane = threadIdx.x;
  int b = bc >> 8;
  const float* pr = proto + (size_t)bc * D_;
  const float* ce = center + (size_t)b * D_;
  float x[8]; float ss = 0.f;
#pragma unroll
  for (int i = 0; i < 8; ++i) { x[i] = pr[lane + 32 * i] - ce[lane + 32 * i]; ss += x[i] * x[i]; }
  for (int m = 1; m < 32; m <<= 1) ss += __shfl_xor(ss, m);
  float inv = 1.0f / fmaxf(sqrtf(ss), 1e-12f);
#pragma unroll
  for (int i = 0; i < 8; ++i) npe[(size_t)bc * D_ + lane + 32 * i] = (__bf16)(x[i] * inv);
}

// normalize(left_output - center) -> f32 nd
__global__ void k_nd(const float* __restrict__ left, const float* __restrict__ center,
                     float* __restrict__ nd) {
  int b = blockIdx.x, lane = threadIdx.x;
  float x[8]; float ss = 0.f;
#pragma unroll
  for (int i = 0; i < 8; ++i) { x[i] = left[b * D_ + lane + 32 * i] - center[b * D_ + lane + 32 * i]; ss += x[i] * x[i]; }
  for (int m = 1; m < 32; m <<= 1) ss += __shfl_xor(ss, m);
  float inv = 1.0f / fmaxf(sqrtf(ss), 1e-12f);
#pragma unroll
  for (int i = 0; i < 8; ++i) nd[b * D_ + lane + 32 * i] = x[i] * inv;
}

__global__ void k_init_rmax(unsigned* __restrict__ rmax) {
  rmax[blockIdx.x * blockDim.x + threadIdx.x] = 0u;   // key 0 < key(any real value)
}

// row-max of a pair of 16x16 result tiles sharing rows (cols n0.. and n0+16..),
// with identity subtraction, folded into global per-row atomicMax
__device__ __forceinline__ void tilepair_rowmax(v8f aL, v8f aR, int mrow, int n0,
                                                unsigned* __restrict__ rmaxRow, int lane) {
  const int c0 = n0 + (lane & 15);
  const int c1 = n0 + 16 + (lane & 15);
  const int h = lane >> 4;
#pragma unroll
  for (int r = 0; r < 8; ++r) {
    const int row = mrow + r + 8 * h;
    float v = fmaxf(aL[r] - (row == c0 ? 1.0f : 0.0f),
                    aR[r] - (row == c1 ? 1.0f : 0.0f));
    v = fmaxf(v, __shfl_xor(v, 1));
    v = fmaxf(v, __shfl_xor(v, 2));
    v = fmaxf(v, __shfl_xor(v, 4));
    v = fmaxf(v, __shfl_xor(v, 8));                 // max over 32 cols of the pair
    if ((lane & 15) == 0) atomicMax(&rmaxRow[row], f2ord(v));
  }
}

// ---------------- cos Gram (WMMA, 32x32 tile / wave) + fused row-max ----------------
__global__ void k_cos(const __bf16* __restrict__ npe, unsigned* __restrict__ rmax) {
  const int lane = threadIdx.x;
  const int b = blockIdx.x >> 6;                    // 8x8 = 64 super-tiles per batch
  const int t = blockIdx.x & 63;
  const int m0 = (t >> 3) * 32, n0 = (t & 7) * 32;
  const __bf16* Nb = npe + (size_t)b * C_ * D_;
  v8f acc00 = {}, acc01 = {}, acc10 = {}, acc11 = {};
  for (int kb = 0; kb < D_; kb += 32) {
    v16bf a0 = frag_a(Nb, D_, m0,      kb, lane);
    v16bf a1 = frag_a(Nb, D_, m0 + 16, kb, lane);
    v16bf b0 = frag_bt(Nb, D_, n0,      kb, lane);
    v16bf b1 = frag_bt(Nb, D_, n0 + 16, kb, lane);
    acc00 = __builtin_amdgcn_wmma_f32_16x16x32_bf16(false, a0, false, b0, (short)0, acc00, false, false);
    acc01 = __builtin_amdgcn_wmma_f32_16x16x32_bf16(false, a0, false, b1, (short)0, acc01, false, false);
    acc10 = __builtin_amdgcn_wmma_f32_16x16x32_bf16(false, a1, false, b0, (short)0, acc10, false, false);
    acc11 = __builtin_amdgcn_wmma_f32_16x16x32_bf16(false, a1, false, b1, (short)0, acc11, false, false);
  }
  unsigned* rmaxB = rmax + (size_t)b * C_;
  tilepair_rowmax(acc00, acc01, m0,      n0, rmaxB, lane);
  tilepair_rowmax(acc10, acc11, m0 + 16, n0, rmaxB, lane);
}

// ---------------- loss2: max over b' of (nd nd^T - I) per row ----------------
__global__ void k_loss2(const float* __restrict__ nd, float* __restrict__ l2b) {
  int b = blockIdx.x, lane = threadIdx.x;
  float mx = -3.402823466e38f;
  for (int b2 = lane; b2 < B_; b2 += 32) {
    float s = 0.f;
    for (int d = 0; d < D_; ++d) s += nd[b * D_ + d] * nd[b2 * D_ + d];
    if (b2 == b) s -= 1.0f;
    mx = fmaxf(mx, s);
  }
  for (int m = 1; m < 32; m <<= 1) mx = fmaxf(mx, __shfl_xor(mx, m));
  if (lane == 0) l2b[b] = mx;
}

// ---------------- final scalar combine ----------------
__global__ void k_final(const float* __restrict__ ceb, const unsigned* __restrict__ rmax,
                        const float* __restrict__ l2b, float* __restrict__ out) {
  __shared__ float red[256];
  int t = threadIdx.x;
  float v = 0.f;
  for (int i = t; i < B_; i += 256) v += ceb[i] * (1.0f / B_);
  float s1 = 0.f;
  for (int i = t; i < B_ * C_; i += 256) s1 += ord2f(rmax[i]);
  v += 0.7f * s1 * (1.0f / (B_ * C_));
  float s2 = 0.f;
  for (int i = t; i < B_; i += 256) s2 += l2b[i];
  v += 0.7f * s2 * (1.0f / B_);
  red[t] = v; __syncthreads();
  for (int s = 128; s > 0; s >>= 1) { if (t < s) red[t] += red[t + s]; __syncthreads(); }
  if (t == 0) out[0] = red[0];
}

// ---------------- host launcher ----------------
extern "C" void kernel_launch(void* const* d_in, const int* in_sizes, int n_in,
                              void* d_out, int out_size, void* d_ws, size_t ws_size,
                              hipStream_t stream) {
  (void)in_sizes; (void)n_in; (void)out_size; (void)ws_size;
  const float* left  = (const float*)d_in[0];   // [128,256]
  const float* drug  = (const float*)d_in[1];   // [128,16,300]
  const float* sem   = (const float*)d_in[2];   // [256,8,256]
  const float* Wq    = (const float*)d_in[3];   // [300,256]
  const float* Wk    = (const float*)d_in[4];   // [256,256]
  const float* Wv    = (const float*)d_in[5];   // [256,256]
  const int*   ids   = (const int*)d_in[6];     // [128]
  float* out = (float*)d_out;

  char* base = (char*)d_ws;
  size_t off = 0;
  auto alloc = [&](size_t bytes) { char* p = base + off; off = (off + bytes + 255) & ~(size_t)255; return p; };
  __bf16*  adrug = (__bf16*)alloc((size_t)MQ_ * DQP_ * 2);
  __bf16*  wqt   = (__bf16*)alloc((size_t)D_ * DQP_ * 2);
  __bf16*  sbf   = (__bf16*)alloc((size_t)MS_ * D_ * 2);
  __bf16*  wkt   = (__bf16*)alloc((size_t)D_ * D_ * 2);
  __bf16*  wvt   = (__bf16*)alloc((size_t)D_ * D_ * 2);
  __bf16*  qb    = (__bf16*)alloc((size_t)MQ_ * D_ * 2);
  __bf16*  kb    = (__bf16*)alloc((size_t)MS_ * D_ * 2);
  float*   vf    = (float*)alloc((size_t)MS_ * D_ * 4);
  float*   abar  = (float*)alloc((size_t)B_ * C_ * 8 * 4);
  float*   proto = (float*)alloc((size_t)B_ * C_ * D_ * 4);
  float*   cen   = (float*)alloc((size_t)B_ * D_ * 4);
  float*   lgt   = (float*)alloc((size_t)B_ * C_ * 4);
  float*   ceb   = (float*)alloc((size_t)B_ * 4);
  __bf16*  npe   = (__bf16*)alloc((size_t)B_ * C_ * D_ * 2);
  float*   nd    = (float*)alloc((size_t)B_ * D_ * 4);
  unsigned* rmax = (unsigned*)alloc((size_t)B_ * C_ * 4);
  float*   l2b   = (float*)alloc((size_t)B_ * 4);

  // Stage 1: bf16 conversions (pad K of drug path 300 -> 320)
  k_cvt_pad  <<<MQ_, 256, 0, stream>>>(drug, adrug, DQ_, DQP_);
  k_cvt_T_pad<<<D_,  256, 0, stream>>>(Wq, wqt, D_, DQ_, DQP_);
  k_cvt_pad  <<<MS_, 256, 0, stream>>>(sem, sbf, D_, D_);
  k_cvt_T_pad<<<D_,  256, 0, stream>>>(Wk, wkt, D_, D_, D_);
  k_cvt_T_pad<<<D_,  256, 0, stream>>>(Wv, wvt, D_, D_, D_);

  // Stage 2: projection GEMMs (WMMA bf16 -> f32 acc), 32x32 tile per wave
  gemm_nt<1><<<(MQ_ / 32) * (D_ / 32), 32, 0, stream>>>(adrug, wqt, qb, D_ / 32, DQP_, DQP_, DQP_, D_);
  gemm_nt<1><<<(MS_ / 32) * (D_ / 32), 32, 0, stream>>>(sbf, wkt, kb, D_ / 32, D_, D_, D_, D_);
  gemm_nt<0><<<(MS_ / 32) * (D_ / 32), 32, 0, stream>>>(sbf, wvt, vf, D_ / 32, D_, D_, D_, D_);

  // Stage 3: fused attention scores -> softmax -> abar (WMMA)
  k_scores_abar<<<B_ * (C_ / 2), 32, 0, stream>>>(kb, qb, abar);

  // Stage 4: proto, center, logits, CE
  k_proto <<<B_ * C_, D_, 0, stream>>>(abar, vf, proto);
  k_center<<<B_, D_, 0, stream>>>(proto, cen);
  k_logits<<<B_, C_, 0, stream>>>(left, proto, lgt);
  k_ce    <<<B_, 32, 0, stream>>>(lgt, ids, ceb);

  // Stage 5: uniformity terms
  k_npe<<<B_ * C_, 32, 0, stream>>>(proto, cen, npe);
  k_nd <<<B_, 32, 0, stream>>>(left, cen, nd);
  k_init_rmax<<<(B_ * C_) / 256, 256, 0, stream>>>(rmax);
  k_cos<<<B_ * 64, 32, 0, stream>>>(npe, rmax);       // WMMA Gram 32x32/wave + fused rowmax
  k_loss2<<<B_, 32, 0, stream>>>(nd, l2b);

  // Stage 6: scalar combine
  k_final<<<1, 256, 0, stream>>>(ceb, rmax, l2b, out);
}
